// BayesLSTM_15152644620990
// MI455X (gfx1250) — matrix operand order repaired and checked by reading, not compiled, not run on hardware
//
#include <hip/hip_runtime.h>
#include <hip/hip_bf16.h>
#include <math.h>

// Problem dims (fixed by the reference)
#define B_  64
#define T_  128
#define IN_ 64
#define H_  256
#define G_  (4 * H_)
#define KSPLIT 8   // split-K factor for the gates GEMM/sampling (deterministic reduction)

typedef _Float16 v16h __attribute__((ext_vector_type(16)));
typedef _Float16 v8h  __attribute__((ext_vector_type(8)));
typedef float    v8f  __attribute__((ext_vector_type(8)));

// ---------------- deterministic Gaussian sampler (splitmix64 + Box-Muller) ---
__device__ __forceinline__ unsigned long long mix64(unsigned long long z) {
    z ^= z >> 30; z *= 0xBF58476D1CE4E5B9ull;
    z ^= z >> 27; z *= 0x94D049BB133111EBull;
    z ^= z >> 31;
    return z;
}

__device__ __forceinline__ float gauss(unsigned int stream, unsigned int idx) {
    unsigned long long z =
        mix64(((unsigned long long)stream << 32) ^
              (0x9E3779B97F4A7C15ull * (unsigned long long)(idx + 1u)));
    unsigned int lo = (unsigned int)z;
    unsigned int hi = (unsigned int)(z >> 32);
    float u1 = (float)(hi >> 8) * 0x1.0p-24f + 0x1.0p-25f;  // (0,1)
    float u2 = (float)(lo >> 8) * 0x1.0p-24f;               // [0,1)
    float r  = sqrtf(-2.0f * __logf(u1));
    return r * __cosf(6.28318530717958647692f * u2);
}

// ---------------- gates_part[s] = [x;h][:,Ks] @ sample(W)[:,Ks]^T (+bias on s==0)
// Grid: (G/64, KSPLIT). Block: 128 threads (4 wave32).
// Each block computes a 64(batch) x 64(gate) partial tile over its K slice.
__global__ __launch_bounds__(128)
void bayes_gates_wmma(const float* __restrict__ ff, int ff_stride, int Kin,
                      const float* __restrict__ hprev,
                      const float* __restrict__ wih_mu, const float* __restrict__ wih_lv,
                      const float* __restrict__ whh_mu, const float* __restrict__ whh_lv,
                      const float* __restrict__ bih_mu, const float* __restrict__ bih_lv,
                      const float* __restrict__ bhh_mu, const float* __restrict__ bhh_lv,
                      float* __restrict__ gates_part, unsigned int seed)
{
    __shared__ alignas(32) _Float16 sA[64 * 32];   // [batch row][k-chunk]
    __shared__ alignas(32) _Float16 sB[64 * 32];   // [gate row ][k-chunk] (= B^T tile)

    const int tid   = threadIdx.x;
    const int lane  = tid & 31;
    const int wv    = tid >> 5;
    const int g0    = blockIdx.x * 64;
    const int split = blockIdx.y;
    const int Ktot  = Kin + H_;

    // K slice for this split (in 32-wide chunks, uneven tail splits may be empty)
    const int nchunks = (Ktot + 31) / 32;
    const int cps     = (nchunks + KSPLIT - 1) / KSPLIT;
    const int kk_beg  = split * cps * 32;
    int kk_end        = (split + 1) * cps * 32;
    if (kk_end > Ktot) kk_end = Ktot;

    v8f acc[4];
#pragma unroll
    for (int n = 0; n < 4; ++n)
#pragma unroll
        for (int i = 0; i < 8; ++i) acc[n][i] = 0.0f;

    for (int kk = kk_beg; kk < kk_end; kk += 32) {
        // ---- stage A: concat(x_t, h_prev) chunk -> f16 LDS (64 x 32)
        for (int e = tid; e < 64 * 32; e += 128) {
            int m = e >> 5, kc = e & 31, k = kk + kc;
            float v = (k < Kin) ? ff[m * ff_stride + k]
                                : hprev[m * H_ + (k - Kin)];
            sA[e] = (_Float16)v;
        }
        // ---- stage B: sample weights (fused reparameterization) -> f16 LDS (64 x 32)
        for (int e = tid; e < 64 * 32; e += 128) {
            int n = e >> 5, kc = e & 31, k = kk + kc;
            int g = g0 + n;
            float mu, lv; unsigned int strm, idx;
            if (k < Kin) {
                idx = (unsigned int)(g * Kin + k);
                mu = wih_mu[idx]; lv = wih_lv[idx]; strm = seed * 4u + 0u;
            } else {
                int kh = k - Kin;
                idx = (unsigned int)(g * H_ + kh);
                mu = whh_mu[idx]; lv = whh_lv[idx]; strm = seed * 4u + 1u;
            }
            float w = mu + __expf(0.5f * lv) * gauss(strm, idx);
            sB[e] = (_Float16)w;
        }
        __syncthreads();

        // ---- A fragment (16x32): lanes 0-15 K=0..15, lanes 16-31 K=16..31
        //      (contiguous, matching B).
        const int k0a = (lane < 16) ? 0 : 16;
        const int m   = wv * 16 + (lane & 15);
        v16h afrag = *(const v16h*)&sA[m * 32 + k0a];

        // ---- B fragments (32x16): lane -> column N=lane%16; lanes 0-15 K=0..15,
        //      lanes 16-31 K=16..31; one contiguous 32B LDS load per fragment.
        const int k0b = (lane < 16) ? 0 : 16;
#pragma unroll
        for (int n = 0; n < 4; ++n) {
            const int col = n * 16 + (lane & 15);
            v16h bfrag = *(const v16h*)&sB[col * 32 + k0b];
            acc[n] = __builtin_amdgcn_wmma_f32_16x16x32_f16(
                false, afrag, false, bfrag, (short)0, acc[n], false, false);
        }
        __syncthreads();
    }

    // ---- store partial (+ sampled biases from split 0 only).
    // C layout: VGPR i -> row i (lanes 0-15) / i+8 (lanes 16-31).
    float* part = gates_part + (size_t)split * (B_ * G_);
    const int rowbase = wv * 16 + ((lane >> 4) << 3);
#pragma unroll
    for (int n = 0; n < 4; ++n) {
        int col = n * 16 + (lane & 15);
        int g   = g0 + col;
        float bias = 0.0f;
        if (split == 0) {
            float bih = bih_mu[g] + __expf(0.5f * bih_lv[g]) * gauss(seed * 4u + 2u, (unsigned int)g);
            float bhh = bhh_mu[g] + __expf(0.5f * bhh_lv[g]) * gauss(seed * 4u + 3u, (unsigned int)g);
            bias = bih + bhh;
        }
#pragma unroll
        for (int i = 0; i < 8; ++i) {
            int mr = rowbase + i;
            part[mr * G_ + g] = acc[n][i] + bias;
        }
    }
}

// ---------------- pointwise LSTM cell (reduces the KSPLIT partials, fixed order)
__global__ __launch_bounds__(256)
void bayes_cell(const float* __restrict__ gates_part,
                float* __restrict__ hbuf, float* __restrict__ cbuf,
                float* __restrict__ out_seq,      // layer1: output (B,T,H); layer0: nullptr
                float* __restrict__ hn, float* __restrict__ cn, int t)
{
    int idx = blockIdx.x * blockDim.x + threadIdx.x;
    if (idx >= B_ * H_) return;
    int b = idx / H_, h = idx % H_;

    float gi = 0.0f, gf = 0.0f, gg = 0.0f, go = 0.0f;
#pragma unroll
    for (int s = 0; s < KSPLIT; ++s) {
        const float* gp = gates_part + (size_t)s * (B_ * G_) + b * G_;
        gi += gp[h];
        gf += gp[H_ + h];
        gg += gp[2 * H_ + h];
        go += gp[3 * H_ + h];
    }

    float c  = cbuf[idx];
    float si = 1.0f / (1.0f + __expf(-gi));
    float sf = 1.0f / (1.0f + __expf(-gf));
    float so = 1.0f / (1.0f + __expf(-go));
    float tg = tanhf(gg);

    float c_new = sf * c + si * tg;
    float h_new = so * tanhf(c_new);

    cbuf[idx] = c_new;
    hbuf[idx] = h_new;
    if (out_seq) out_seq[(b * T_ + t) * H_ + h] = h_new;
    if (t == T_ - 1) { hn[idx] = h_new; cn[idx] = c_new; }
}

__global__ void zero_f32(float* __restrict__ p, int n) {
    int i = blockIdx.x * blockDim.x + threadIdx.x;
    if (i < n) p[i] = 0.0f;
}

// ---------------- host-side orchestration -------------------------------------
extern "C" void kernel_launch(void* const* d_in, const int* in_sizes, int n_in,
                              void* d_out, int out_size, void* d_ws, size_t ws_size,
                              hipStream_t stream)
{
    const float* x        = (const float*)d_in[0];
    const float* wih_mu0  = (const float*)d_in[1];
    const float* wih_lv0  = (const float*)d_in[2];
    const float* whh_mu0  = (const float*)d_in[3];
    const float* whh_lv0  = (const float*)d_in[4];
    const float* bih_mu0  = (const float*)d_in[5];
    const float* bih_lv0  = (const float*)d_in[6];
    const float* bhh_mu0  = (const float*)d_in[7];
    const float* bhh_lv0  = (const float*)d_in[8];
    const float* wih_mu1  = (const float*)d_in[9];
    const float* wih_lv1  = (const float*)d_in[10];
    const float* whh_mu1  = (const float*)d_in[11];
    const float* whh_lv1  = (const float*)d_in[12];
    const float* bih_mu1  = (const float*)d_in[13];
    const float* bih_lv1  = (const float*)d_in[14];
    const float* bhh_mu1  = (const float*)d_in[15];
    const float* bhh_lv1  = (const float*)d_in[16];

    float* ws         = (float*)d_ws;
    float* gates_part = ws;                                 // KSPLIT*B*G f32 (2 MB)
    float* h0         = gates_part + (size_t)KSPLIT * B_ * G_;
    float* c0         = h0 + B_ * H_;
    float* h1         = c0 + B_ * H_;
    float* c1         = h1 + B_ * H_;

    float* out     = (float*)d_out;                         // (B,T,H)
    float* hn_base = out + (size_t)B_ * T_ * H_;            // (2,B,H)
    float* cn_base = hn_base + 2 * B_ * H_;                 // (2,B,H)

    // zero h0,c0,h1,c1 (contiguous; ws is poisoned by the harness)
    zero_f32<<<(4 * B_ * H_ + 255) / 256, 256, 0, stream>>>(h0, 4 * B_ * H_);

    const dim3 ggrid(G_ / 64, KSPLIT);
    const int  cell_blocks = (B_ * H_ + 255) / 256;

    for (int t = 0; t < T_; ++t) {
        unsigned int seed0 = (unsigned int)(2 * t + 0) + 1u;
        unsigned int seed1 = (unsigned int)(2 * t + 1) + 1u;

        // layer 0: ff input = x[:, t, :]  (element (b,k) at (x + t*IN)[b*T*IN + k])
        bayes_gates_wmma<<<ggrid, 128, 0, stream>>>(
            x + (size_t)t * IN_, T_ * IN_, IN_, h0,
            wih_mu0, wih_lv0, whh_mu0, whh_lv0,
            bih_mu0, bih_lv0, bhh_mu0, bhh_lv0, gates_part, seed0);
        bayes_cell<<<cell_blocks, 256, 0, stream>>>(
            gates_part, h0, c0, nullptr, hn_base, cn_base, t);

        // layer 1: ff input = h0 (just produced for time t)
        bayes_gates_wmma<<<ggrid, 128, 0, stream>>>(
            h0, H_, H_, h1,
            wih_mu1, wih_lv1, whh_mu1, whh_lv1,
            bih_mu1, bih_lv1, bhh_mu1, bhh_lv1, gates_part, seed1);
        bayes_cell<<<cell_blocks, 256, 0, stream>>>(
            gates_part, h1, c1, out, hn_base + B_ * H_, cn_base + B_ * H_, t);
    }
}
